// VectorQuantizer2_79250736546256
// MI455X (gfx1250) — matrix operand order repaired and checked by reading, not compiled, not run on hardware
//
#include <hip/hip_runtime.h>
#include <math.h>

// ---------------------------------------------------------------------------
// VQ (VectorQuantizer2): N=8192 vectors (D=1024) vs K=8192 codes.
// Distance GEMM done with v_wmma_f32_16x16x32_bf16 using bf16 hi/lo split
// (3 WMMAs per K-tile step) for ~fp32 accuracy on the argmin scores.
// ---------------------------------------------------------------------------

typedef __bf16  bf16;
typedef __bf16  v16bf __attribute__((ext_vector_type(16)));
typedef float   v8f   __attribute__((ext_vector_type(8)));

#define N_ROWS   8192            // 8 * 32 * 32
#define DIM      1024
#define K_EMB    8192
#define M_BLK    64              // z rows per workgroup (256KB LDS staging)
#define NTOT     (8u * 1024u * 1024u * 8u)   // 8*1024*1024*8 = 67108864

// workspace byte offsets
#define OFF_ZHI    ((size_t)0)
#define OFF_ZLO    ((size_t)16 << 20)
#define OFF_EHI    ((size_t)32 << 20)
#define OFF_ELO    ((size_t)48 << 20)
#define OFF_ENORM  ((size_t)64 << 20)            // 8192 floats
#define OFF_COUNTS (((size_t)64 << 20) + 32768)  // 8192 uints
#define OFF_LOSS   (((size_t)64 << 20) + 65536)  // 1 float
#define OFF_IDX    (((size_t)64 << 20) + 65600)  // 8192 ints

__device__ __forceinline__ void split_bf16(float v, bf16& h, bf16& l) {
  h = (bf16)v;
  l = (bf16)(v - (float)h);
}

// ---------------------------------------------------------------------------
// Kernel: transpose z [b,c,h,w] -> z_flat [n=b*1024+h*32+w, c], bf16 hi/lo.
// 32x32 LDS tile transpose so both sides are coalesced.
// ---------------------------------------------------------------------------
__global__ void prep_z_kernel(const float* __restrict__ z,
                              bf16* __restrict__ zhi, bf16* __restrict__ zlo) {
  __shared__ float tile[32][33];
  const int tid = threadIdx.x;
  const int tx = tid & 31, ty = tid >> 5;       // 32 x 8
  const int ht = blockIdx.x;                    // hw tile (32)
  const int ct = blockIdx.y;                    // c  tile (32)
  const int b  = blockIdx.z;                    // batch (8)
#pragma unroll
  for (int j = 0; j < 4; ++j) {
    int cc = ty + j * 8;
    tile[cc][tx] = z[(size_t)b * 1048576 + (size_t)(ct * 32 + cc) * 1024
                     + ht * 32 + tx];
  }
  __syncthreads();
#pragma unroll
  for (int j = 0; j < 4; ++j) {
    int nn = ty + j * 8;
    int n  = b * 1024 + ht * 32 + nn;
    int c  = ct * 32 + tx;
    bf16 h, l;
    split_bf16(tile[tx][nn], h, l);
    zhi[(size_t)n * DIM + c] = h;
    zlo[(size_t)n * DIM + c] = l;
  }
}

// ---------------------------------------------------------------------------
// Kernel: split codebook into bf16 hi/lo planes (same row-major layout).
// ---------------------------------------------------------------------------
__global__ void prep_e_kernel(const float* __restrict__ e,
                              bf16* __restrict__ ehi, bf16* __restrict__ elo) {
  size_t i = (size_t)blockIdx.x * blockDim.x + threadIdx.x;
  bf16 h, l;
  split_bf16(e[i], h, l);
  ehi[i] = h;
  elo[i] = l;
}

// ---------------------------------------------------------------------------
// Kernel: per-code squared norm |e_k|^2 (fp32 exact).
// ---------------------------------------------------------------------------
__global__ void enorm_kernel(const float* __restrict__ e,
                             float* __restrict__ enorm) {
  const int k = blockIdx.x;
  const int tid = threadIdx.x;
  const float* row = e + (size_t)k * DIM;
  float s = 0.f;
  for (int i = tid; i < DIM; i += 256) { float v = row[i]; s += v * v; }
  for (int off = 16; off > 0; off >>= 1) s += __shfl_down(s, off);
  __shared__ float red[8];
  if ((tid & 31) == 0) red[tid >> 5] = s;
  __syncthreads();
  if (tid == 0) {
    float t = 0.f;
#pragma unroll
    for (int i = 0; i < 8; ++i) t += red[i];
    enorm[k] = t;
  }
}

// ---------------------------------------------------------------------------
// Main kernel: per 64-row block, scan all 8192 codes with WMMA and keep a
// running min of (|e|^2 - 2 z.e).  8 waves = 4 M-tiles x 2 K-slices.
// ---------------------------------------------------------------------------
__global__ __launch_bounds__(256)
void vq_argmin_kernel(const bf16* __restrict__ zhi, const bf16* __restrict__ zlo,
                      const bf16* __restrict__ ehi, const bf16* __restrict__ elo,
                      const float* __restrict__ enorm,
                      int* __restrict__ idx_out, float* __restrict__ idx_f_out,
                      unsigned* __restrict__ counts) {
  __shared__ bf16 lds_zhi[M_BLK * DIM];   // 128 KB
  __shared__ bf16 lds_zlo[M_BLK * DIM];   // 128 KB
  __shared__ float red_v[8][16];
  __shared__ int   red_k[8][16];

  const int tid  = threadIdx.x;
  const int lane = tid & 31;
  const int wave = tid >> 5;
  const int wg   = blockIdx.x;

  // ---- stage this block's 64 z-rows (hi+lo) into LDS, vectorized b128 ----
  {
    const uint4* gh = (const uint4*)(zhi + (size_t)wg * M_BLK * DIM);
    const uint4* gl = (const uint4*)(zlo + (size_t)wg * M_BLK * DIM);
    uint4* lh = (uint4*)lds_zhi;
    uint4* ll = (uint4*)lds_zlo;
    for (int i = tid; i < (M_BLK * DIM) / 8; i += 256) {
      lh[i] = gh[i];
      ll[i] = gl[i];
    }
  }
  __syncthreads();

  const int mtile  = wave >> 1;      // which 16-row tile (0..3)
  const int kslice = wave & 1;       // which half of the 512 k-tiles
  const int col    = lane & 15;
  const int half   = lane >> 4;

  // A fragment per-lane base (16-bit A 16x32 layout):
  // lanes 0-15: rows 0-15, K {d0..d0+7, d0+16..d0+23}; lanes 16-31: +8 shift.
  const bf16* arow = lds_zhi + (size_t)(mtile * 16 + col) * DIM + half * 8;
  const bf16* alow = lds_zlo + (size_t)(mtile * 16 + col) * DIM + half * 8;

  float minv[8];
  int   mink[8];
#pragma unroll
  for (int i = 0; i < 8; ++i) { minv[i] = 3.4e38f; mink[i] = 0; }

  for (int kt = kslice; kt < K_EMB / 16; kt += 2) {
    // Memory clobber: stops the compiler from hoisting the (k-invariant)
    // LDS A-fragment loads out of this loop, which previously ballooned to
    // a 512-VGPR panel and spilled to scratch in the hot loop.
    asm volatile("" ::: "memory");

    const int kcol = kt * 16 + col;
    // B fragment per-lane base: lane holds K(=d) {d0..d0+7, d0+16..d0+23} of
    // codebook row kcol; lanes 16-31: +8 shift (same layout as A).
    const bf16* brh = ehi + (size_t)kcol * DIM + half * 8;
    const bf16* brl = elo + (size_t)kcol * DIM + half * 8;

    v8f acc = {0.f, 0.f, 0.f, 0.f, 0.f, 0.f, 0.f, 0.f};

#pragma unroll
    for (int d0 = 0; d0 < DIM; d0 += 32) {
      v16bf ah, al, bh, bl;
      ((uint4*)&ah)[0] = *(const uint4*)(arow + d0);
      ((uint4*)&ah)[1] = *(const uint4*)(arow + d0 + 16);
      ((uint4*)&al)[0] = *(const uint4*)(alow + d0);
      ((uint4*)&al)[1] = *(const uint4*)(alow + d0 + 16);
      ((uint4*)&bh)[0] = *(const uint4*)(brh + d0);
      ((uint4*)&bh)[1] = *(const uint4*)(brh + d0 + 16);
      ((uint4*)&bl)[0] = *(const uint4*)(brl + d0);
      ((uint4*)&bl)[1] = *(const uint4*)(brl + d0 + 16);

      acc = __builtin_amdgcn_wmma_f32_16x16x32_bf16(false, ah, false, bh,
                                                    (short)0, acc, false, false);
      acc = __builtin_amdgcn_wmma_f32_16x16x32_bf16(false, ah, false, bl,
                                                    (short)0, acc, false, false);
      acc = __builtin_amdgcn_wmma_f32_16x16x32_bf16(false, al, false, bh,
                                                    (short)0, acc, false, false);
    }

    const float en = enorm[kcol];
#pragma unroll
    for (int i = 0; i < 8; ++i) {
      float d = en - 2.f * acc[i];
      if (d < minv[i]) { minv[i] = d; mink[i] = kcol; }
    }
  }

  // ---- butterfly min across the 16 column lanes of each half ----
  for (int off = 1; off < 16; off <<= 1) {
#pragma unroll
    for (int i = 0; i < 8; ++i) {
      float ov = __shfl_xor(minv[i], off);
      int   ok = __shfl_xor(mink[i], off);
      if (ov < minv[i] || (ov == minv[i] && ok < mink[i])) {
        minv[i] = ov; mink[i] = ok;
      }
    }
  }
  // lane 0 holds rows 0..7 of the tile, lane 16 holds rows 8..15
  if (col == 0) {
#pragma unroll
    for (int i = 0; i < 8; ++i) {
      red_v[wave][half * 8 + i] = minv[i];
      red_k[wave][half * 8 + i] = mink[i];
    }
  }
  __syncthreads();

  // ---- cross-wave (2 K-slices per M-tile) final reduce + emit ----
  if (tid < 64) {
    const int mt = tid >> 4, r = tid & 15;
    const int w0 = mt * 2, w1 = mt * 2 + 1;
    float bv = red_v[w0][r];
    int   bk = red_k[w0][r];
    float v1 = red_v[w1][r];
    int   k1 = red_k[w1][r];
    if (v1 < bv || (v1 == bv && k1 < bk)) { bv = v1; bk = k1; }
    const int row = wg * M_BLK + mt * 16 + r;
    idx_out[row]   = bk;
    idx_f_out[row] = (float)bk;
    atomicAdd(&counts[bk], 1u);
  }
}

// ---------------------------------------------------------------------------
// Kernel: z_q_out[b,c,h,w] = E[idx[n], c]  (+ accumulate sum((zq-z)^2)).
// ---------------------------------------------------------------------------
__global__ void gather_loss_kernel(const float* __restrict__ z,
                                   const float* __restrict__ emb,
                                   const int* __restrict__ idxs,
                                   float* __restrict__ out,
                                   float* __restrict__ loss_acc) {
  const unsigned flat = blockIdx.x * 256u + threadIdx.x;   // out layout b,c,h,w
  const unsigned b  = flat >> 20;
  const unsigned rem = flat & 1048575u;
  const unsigned c  = rem >> 10;
  const unsigned hw = rem & 1023u;
  const unsigned n  = (b << 10) | hw;
  const int k = idxs[n];
  const float e  = emb[(size_t)k * DIM + c];
  const float zv = z[flat];
  out[flat] = e;
  float d = e - zv;
  float s = d * d;
  for (int off = 16; off > 0; off >>= 1) s += __shfl_down(s, off);
  __shared__ float red[8];
  const int tid = threadIdx.x;
  if ((tid & 31) == 0) red[tid >> 5] = s;
  __syncthreads();
  if (tid == 0) {
    float t = 0.f;
#pragma unroll
    for (int i = 0; i < 8; ++i) t += red[i];
    atomicAdd(loss_acc, t);
  }
}

// ---------------------------------------------------------------------------
// Kernel: final scalars -> loss, perplexity.
// ---------------------------------------------------------------------------
__global__ void scalars_kernel(const unsigned* __restrict__ counts,
                               const float* __restrict__ loss_acc,
                               float* __restrict__ out_scalars) {
  const int tid = threadIdx.x;
  float s = 0.f;
  for (int k = tid; k < K_EMB; k += 256) {
    float p = (float)counts[k] * (1.0f / (float)N_ROWS);
    s += p * logf(p + 1e-10f);
  }
  for (int off = 16; off > 0; off >>= 1) s += __shfl_down(s, off);
  __shared__ float red[8];
  if ((tid & 31) == 0) red[tid >> 5] = s;
  __syncthreads();
  if (tid == 0) {
    float t = 0.f;
#pragma unroll
    for (int i = 0; i < 8; ++i) t += red[i];
    out_scalars[0] = loss_acc[0] * 1.25f / (float)NTOT;  // (1+beta)*MSE
    out_scalars[1] = expf(-t);                            // perplexity
  }
}

// ---------------------------------------------------------------------------
extern "C" void kernel_launch(void* const* d_in, const int* in_sizes, int n_in,
                              void* d_out, int out_size, void* d_ws, size_t ws_size,
                              hipStream_t stream) {
  (void)in_sizes; (void)n_in; (void)out_size; (void)ws_size;
  const float* z   = (const float*)d_in[0];   // [8,1024,32,32]
  const float* emb = (const float*)d_in[1];   // [8192,1024]
  float* out = (float*)d_out;                 // [zq | loss | perp | idx]

  char* ws = (char*)d_ws;
  bf16*     zhi    = (bf16*)(ws + OFF_ZHI);
  bf16*     zlo    = (bf16*)(ws + OFF_ZLO);
  bf16*     ehi    = (bf16*)(ws + OFF_EHI);
  bf16*     elo    = (bf16*)(ws + OFF_ELO);
  float*    enorm  = (float*)(ws + OFF_ENORM);
  unsigned* counts = (unsigned*)(ws + OFF_COUNTS);
  float*    lossa  = (float*)(ws + OFF_LOSS);
  int*      idxs   = (int*)(ws + OFF_IDX);

  // zero counts + loss accumulator (adjacent in ws)
  hipMemsetAsync(ws + OFF_COUNTS, 0, 32768 + 64, stream);

  // prep: transpose/split z, split codebook, code norms
  prep_z_kernel<<<dim3(32, 32, 8), 256, 0, stream>>>(z, zhi, zlo);
  prep_e_kernel<<<(K_EMB * DIM) / 256, 256, 0, stream>>>(emb, ehi, elo);
  enorm_kernel<<<K_EMB, 256, 0, stream>>>(emb, enorm);

  // main WMMA distance + argmin
  float* out_idx_f = out + NTOT + 2;
  vq_argmin_kernel<<<N_ROWS / M_BLK, 256, 0, stream>>>(
      zhi, zlo, ehi, elo, enorm, idxs, out_idx_f, counts);

  // gather z_q back to NCHW + loss accumulation
  gather_loss_kernel<<<NTOT / 256, 256, 0, stream>>>(z, emb, idxs, out, lossa);

  // scalars
  scalars_kernel<<<1, 256, 0, stream>>>(counts, lossa, out + NTOT);
}
